// CrossAttention_31791347925417
// MI455X (gfx1250) — matrix operand order repaired and checked by reading, not compiled, not run on hardware
//
#include <hip/hip_runtime.h>

// CrossAttention (linear-attention style, D=C/H=64) for MI455X / gfx1250.
// All matmuls via V_WMMA_F32_16X16X32_F16 (f16 in, f32 accumulate).
// See header analysis: HBM roofline ~17us; f16 WMMA keeps compute under it.

typedef _Float16 half_t;
typedef __attribute__((ext_vector_type(16))) _Float16 v16h;
typedef __attribute__((ext_vector_type(8)))  _Float16 v8h;
typedef __attribute__((ext_vector_type(8)))  float    v8f;

#define B_  8
#define N_  4096
#define C_  512
#define H_  8
#define D_  64
#define SCALE_ 0.125f   // D^-0.5

__device__ __forceinline__ v8f wmma16(v16h a, v16h b, v8f c) {
  return __builtin_amdgcn_wmma_f32_16x16x32_f16(false, a, false, b, (short)0, c,
                                                false, false);
}

// A-fragment (16x32, row-major source): lane = row (lo),
// 16 contiguous K values. Caller passes rowPtr + kb + hi*16.
__device__ __forceinline__ v16h ldA(const half_t* p) {
  union { v16h v; v8h h[2]; } u;
  u.h[0] = *(const v8h*)(p);
  u.h[1] = *(const v8h*)(p + 8);
  return u.v;
}
// B-fragment (32x16, from B^T stored row-major): lane = column (lo),
// 16 contiguous K values. Caller passes colPtr + kb + hi*16.
__device__ __forceinline__ v16h ldB(const half_t* p) {
  union { v16h v; v8h h[2]; } u;
  u.h[0] = *(const v8h*)(p);
  u.h[1] = *(const v8h*)(p + 8);
  return u.v;
}

// ---------------------------------------------------------------- f32 -> f16
__global__ __launch_bounds__(256)
void ca_cvt_f16(const float* __restrict__ in, half_t* __restrict__ out, size_t n) {
  size_t i = ((size_t)blockIdx.x * blockDim.x + threadIdx.x) * 8;
  if (i + 8 <= n) {
    float4 f0 = *(const float4*)(in + i);
    float4 f1 = *(const float4*)(in + i + 4);
    v8h p;
    p[0] = (half_t)f0.x; p[1] = (half_t)f0.y; p[2] = (half_t)f0.z; p[3] = (half_t)f0.w;
    p[4] = (half_t)f1.x; p[5] = (half_t)f1.y; p[6] = (half_t)f1.z; p[7] = (half_t)f1.w;
    *(v8h*)(out + i) = p;
  }
}

// --------------------------------------------- kv projection, transposed out
// kv[m, j] = sum_c X16[m, c] * W16[j, c]   (W is [2C, C] = B^T, perfect layout)
// j <  512 -> kT[b][h][d][n] = kv   (h=j/64, d=j%64)
// j >= 512 -> vT[b][h][e][n] = kv
// Workgroup: 4 waves (2x2), 64 rows x 128 cols. Wave: 32x64 (2x4 WMMA tiles).
__global__ __launch_bounds__(128)
void ca_gemm_kv(const half_t* __restrict__ X, const half_t* __restrict__ W,
                half_t* __restrict__ kT, half_t* __restrict__ vT) {
  const int wave = threadIdx.x >> 5;
  const int lane = threadIdx.x & 31;
  const int lo = lane & 15, hi = lane >> 4;
  const int m0 = blockIdx.x * 64 + (wave >> 1) * 32;   // 64 | 4096: no batch cross
  const int j0 = blockIdx.y * 128 + (wave & 1) * 64;

  const half_t* arow0 = X + (size_t)(m0 + lo) * C_;
  const half_t* arow1 = X + (size_t)(m0 + 16 + lo) * C_;
  const half_t* brow0 = W + (size_t)(j0 +  0 + lo) * C_;
  const half_t* brow1 = W + (size_t)(j0 + 16 + lo) * C_;
  const half_t* brow2 = W + (size_t)(j0 + 32 + lo) * C_;
  const half_t* brow3 = W + (size_t)(j0 + 48 + lo) * C_;

  v8f acc[2][4] = {};
  for (int kb = 0; kb < C_; kb += 32) {
    v16h a0 = ldA(arow0 + kb + hi * 16);
    v16h a1 = ldA(arow1 + kb + hi * 16);
    v16h b0 = ldB(brow0 + kb + hi * 16);
    v16h b1 = ldB(brow1 + kb + hi * 16);
    v16h b2 = ldB(brow2 + kb + hi * 16);
    v16h b3 = ldB(brow3 + kb + hi * 16);
    acc[0][0] = wmma16(a0, b0, acc[0][0]);
    acc[0][1] = wmma16(a0, b1, acc[0][1]);
    acc[0][2] = wmma16(a0, b2, acc[0][2]);
    acc[0][3] = wmma16(a0, b3, acc[0][3]);
    acc[1][0] = wmma16(a1, b0, acc[1][0]);
    acc[1][1] = wmma16(a1, b1, acc[1][1]);
    acc[1][2] = wmma16(a1, b2, acc[1][2]);
    acc[1][3] = wmma16(a1, b3, acc[1][3]);
  }

  // D-matrix fragment: lane owns column j, rows M = hi*8 + r (8 consecutive m)
  // -> in the transposed [.,.,d,n] layout this is one contiguous 16B store.
  const int bidx = m0 >> 12;            // m / 4096
  const int nloc = (m0 & (N_ - 1)) + hi * 8;
#pragma unroll
  for (int mt = 0; mt < 2; ++mt) {
#pragma unroll
    for (int t = 0; t < 4; ++t) {
      int j = j0 + t * 16 + lo;
      half_t* dst;
      int hh, cc;
      if (j < C_) { hh = j >> 6; cc = j & 63; dst = kT; }
      else        { int j2 = j - C_; hh = j2 >> 6; cc = j2 & 63; dst = vT; }
      dst += ((size_t)(bidx * H_ + hh) * D_ + cc) * N_ + nloc + mt * 16;
      v8h pk;
#pragma unroll
      for (int r = 0; r < 8; ++r) pk[r] = (half_t)acc[mt][t][r];
      *(v8h*)dst = pk;
    }
  }
}

// ------------------------------------- ctx = softmax_d( K^T V * SCALE ), fused
// One block per (b,h). 4 waves split N=4096; partials reduced in LDS; softmax
// over d per column e; output ctxT[e][d] (f16) so the next GEMM's B-frags are
// contiguous.
__global__ __launch_bounds__(128)
void ca_ctx_softmax(const half_t* __restrict__ kT, const half_t* __restrict__ vT,
                    half_t* __restrict__ ctxT) {
  __shared__ float red[4][D_][D_];   // 64 KB of the 320 KB WGP LDS
  const int wave = threadIdx.x >> 5;
  const int lane = threadIdx.x & 31;
  const int lo = lane & 15, hi = lane >> 4;
  const half_t* kb = kT + (size_t)blockIdx.x * D_ * N_;
  const half_t* vb = vT + (size_t)blockIdx.x * D_ * N_;

  const half_t* arow[4];
  const half_t* brow[4];
#pragma unroll
  for (int t = 0; t < 4; ++t) {
    arow[t] = kb + (size_t)(t * 16 + lo) * N_;   // A row d (K^T is kT row-major)
    brow[t] = vb + (size_t)(t * 16 + lo) * N_;   // B^T row e (vT row-major)
  }

  v8f acc[4][4] = {};
  const int n0 = wave * (N_ / 4);
  for (int nb = n0; nb < n0 + N_ / 4; nb += 32) {
    v16h a[4], b[4];
#pragma unroll
    for (int t = 0; t < 4; ++t) a[t] = ldA(arow[t] + nb + hi * 16);
#pragma unroll
    for (int t = 0; t < 4; ++t) b[t] = ldB(brow[t] + nb + hi * 16);
#pragma unroll
    for (int dt = 0; dt < 4; ++dt)
#pragma unroll
      for (int et = 0; et < 4; ++et)
        acc[dt][et] = wmma16(a[dt], b[et], acc[dt][et]);
  }

#pragma unroll
  for (int dt = 0; dt < 4; ++dt)
#pragma unroll
    for (int et = 0; et < 4; ++et)
#pragma unroll
      for (int r = 0; r < 8; ++r)
        red[wave][dt * 16 + hi * 8 + r][et * 16 + lo] = acc[dt][et][r];
  __syncthreads();

  float* rp = &red[0][0][0];
  for (int idx = threadIdx.x; idx < D_ * D_; idx += 128) {
    float s = rp[idx] + rp[D_ * D_ + idx] + rp[2 * D_ * D_ + idx] + rp[3 * D_ * D_ + idx];
    rp[idx] = s * SCALE_;
  }
  __syncthreads();

  if (threadIdx.x < D_) {
    const int e = threadIdx.x;       // column-softmax over d; bank-conflict free
    float mx = -1e30f;
    for (int d = 0; d < D_; ++d) mx = fmaxf(mx, rp[d * D_ + e]);
    float sum = 0.f;
    for (int d = 0; d < D_; ++d) sum += __expf(rp[d * D_ + e] - mx);
    float inv = 1.f / sum;
    half_t* dst = ctxT + ((size_t)blockIdx.x * D_ + e) * D_;   // row e of ctx^T
    for (int d = 0; d < D_; ++d)
      dst[d] = (half_t)(__expf(rp[d * D_ + e] - mx) * inv);
  }
}

// --------------------------------------------------- out = X_h @ ctx  (f32 out)
// Block: 8 waves, each wave 64 rows x full 64 cols (4x4 WMMA tiles, K=64).
__global__ __launch_bounds__(256)
void ca_out_gemm(const half_t* __restrict__ A16, const half_t* __restrict__ ctxT,
                 float* __restrict__ out) {
  const int bh = blockIdx.y;
  const int b = bh >> 3, h = bh & 7;
  const int wave = threadIdx.x >> 5;
  const int lane = threadIdx.x & 31;
  const int lo = lane & 15, hi = lane >> 4;
  const int n0 = blockIdx.x * 512 + wave * 64;

  const half_t* abase = A16 + (size_t)b * N_ * C_ + (size_t)h * D_;
  const half_t* bbase = ctxT + (size_t)bh * D_ * D_;

  const half_t* arow[4];
  const half_t* brow[4];
#pragma unroll
  for (int t = 0; t < 4; ++t) {
    arow[t] = abase + (size_t)(n0 + t * 16 + lo) * C_;
    brow[t] = bbase + (size_t)(t * 16 + lo) * D_;   // ctx^T row e: contiguous d
  }

  v8f acc[4][4] = {};
#pragma unroll
  for (int kb = 0; kb < D_; kb += 32) {
    v16h a[4], bfr[4];
#pragma unroll
    for (int t = 0; t < 4; ++t) a[t] = ldA(arow[t] + kb + hi * 16);
#pragma unroll
    for (int t = 0; t < 4; ++t) bfr[t] = ldB(brow[t] + kb + hi * 16);
#pragma unroll
    for (int nt = 0; nt < 4; ++nt)
#pragma unroll
      for (int et = 0; et < 4; ++et)
        acc[nt][et] = wmma16(a[nt], bfr[et], acc[nt][et]);
  }

  const size_t outbase = (size_t)b * N_ * C_ + (size_t)h * D_;
#pragma unroll
  for (int nt = 0; nt < 4; ++nt)
#pragma unroll
    for (int et = 0; et < 4; ++et)
#pragma unroll
      for (int r = 0; r < 8; ++r) {
        int n = n0 + nt * 16 + hi * 8 + r;
        out[outbase + (size_t)n * C_ + et * 16 + lo] = acc[nt][et][r];
      }
}

extern "C" void kernel_launch(void* const* d_in, const int* in_sizes, int n_in,
                              void* d_out, int out_size, void* d_ws, size_t ws_size,
                              hipStream_t stream) {
  const float* x  = (const float*)d_in[0];
  const float* bl = (const float*)d_in[1];
  const float* W1 = (const float*)d_in[2];
  const float* W2 = (const float*)d_in[3];
  float* out = (float*)d_out;

  const size_t nXC = (size_t)B_ * N_ * C_;      // 16,777,216
  const size_t nW  = (size_t)2 * C_ * C_;       // 524,288
  const size_t nKT = (size_t)B_ * H_ * D_ * N_; // 16,777,216
  const size_t nCT = (size_t)B_ * H_ * D_ * D_; // 262,144

  half_t* p   = (half_t*)d_ws;                  // ~205 MB total workspace
  half_t* x16 = p; p += nXC;
  half_t* b16 = p; p += nXC;
  half_t* w1h = p; p += nW;
  half_t* w2h = p; p += nW;
  half_t* kT1 = p; p += nKT;
  half_t* vT1 = p; p += nKT;
  half_t* kT2 = p; p += nKT;
  half_t* vT2 = p; p += nKT;
  half_t* cT1 = p; p += nCT;
  half_t* cT2 = p; p += nCT;

  ca_cvt_f16<<<dim3(nXC / 2048), 256, 0, stream>>>(x,  x16, nXC);
  ca_cvt_f16<<<dim3(nXC / 2048), 256, 0, stream>>>(bl, b16, nXC);
  ca_cvt_f16<<<dim3(nW  / 2048), 256, 0, stream>>>(W1, w1h, nW);
  ca_cvt_f16<<<dim3(nW  / 2048), 256, 0, stream>>>(W2, w2h, nW);

  ca_gemm_kv<<<dim3(B_ * N_ / 64, 2 * C_ / 128), 128, 0, stream>>>(x16, w1h, kT1, vT1);
  ca_gemm_kv<<<dim3(B_ * N_ / 64, 2 * C_ / 128), 128, 0, stream>>>(b16, w2h, kT2, vT2);

  ca_ctx_softmax<<<dim3(B_ * H_), 128, 0, stream>>>(kT1, vT1, cT1);
  ca_ctx_softmax<<<dim3(B_ * H_), 128, 0, stream>>>(kT2, vT2, cT2);

  // out_x = x_h @ ctx2 ; out_b = blood_h @ ctx1  (concatenated in d_out)
  ca_out_gemm<<<dim3(N_ / 512, B_ * H_), 256, 0, stream>>>(x16, cT2, out);
  ca_out_gemm<<<dim3(N_ / 512, B_ * H_), 256, 0, stream>>>(b16, cT1, out + nXC);
}
